// DroneDetectorMamba_83339545411729
// MI455X (gfx1250) — matrix operand order repaired and checked by reading, not compiled, not run on hardware
//
#include <hip/hip_runtime.h>
#include <stdint.h>

typedef _Float16 h16;
typedef __attribute__((ext_vector_type(16))) _Float16 v16h;
typedef __attribute__((ext_vector_type(8)))  _Float16 v8h;
typedef __attribute__((ext_vector_type(2)))  _Float16 h2v;
typedef __attribute__((ext_vector_type(8)))  float    v8f;
typedef __attribute__((ext_vector_type(4)))  int      v4i;

#define LN_EPS 1e-5f

// ---- optional CDNA5 async-to-LDS staging (guarded: falls back cleanly) ----
#if defined(__has_builtin)
#if __has_builtin(__builtin_amdgcn_global_load_async_to_lds_b128)
#define HAVE_ASYNC_LDS 1
#endif
#endif
#ifndef HAVE_ASYNC_LDS
#define HAVE_ASYNC_LDS 0
#endif

#if HAVE_ASYNC_LDS
typedef __attribute__((address_space(1))) v4i* gbl_v4i_p;   // global int4*
typedef __attribute__((address_space(3))) v4i* lds_v4i_p;   // LDS int4*

__device__ __forceinline__ void async_copy16(const _Float16* gsrc, _Float16* ldst) {
  // generic -> typed generic -> explicit addrspace cast (C-style handles both)
  v4i* gs = (v4i*)gsrc;
  v4i* ls = (v4i*)ldst;
  __builtin_amdgcn_global_load_async_to_lds_b128((gbl_v4i_p)gs, (lds_v4i_p)ls, 0, 0);
}

__device__ __forceinline__ void async_wait0() {
#if __has_builtin(__builtin_amdgcn_s_wait_asynccnt)
  __builtin_amdgcn_s_wait_asynccnt(0);
#else
  asm volatile("s_wait_asynccnt 0x0" ::: "memory");
#endif
}
#endif

// ---------------------------------------------------------------------------
// WMMA fragment loaders (wave32, v_wmma_f32_16x16x32_f16 layouts per ISA 7.12.2)
// ---------------------------------------------------------------------------
__device__ __forceinline__ v16h frag_a_lds(const _Float16* base, int ld) {
  int lane = threadIdx.x & 31;
  int row = lane & 15, hi = lane >> 4;
  v16h a;
#pragma unroll
  for (int p = 0; p < 8; ++p) {
    // VGPR p: lanes0-15 K = {2p,2p+1} (p<4) or {2p+8,2p+9} (p>=4); lanes16-31: +8
    int k = ((p < 4) ? (2 * p) : (2 * p + 8)) + 8 * hi;
    h2v t = *(const h2v*)(base + row * ld + k);
    a[2 * p] = t[0];
    a[2 * p + 1] = t[1];
  }
  return a;
}

// B tile stored pair-of-k interleaved: element (k,n) at Bs2[(k>>1)*132 + n*2 + (k&1)]
// -> every fragment pair is one aligned 32-bit LDS load.
__device__ __forceinline__ v16h frag_b_pair(const _Float16* Bs2, int ncol0) {
  int lane = threadIdx.x & 31;
  int col = (lane & 15) + ncol0;
  int hi = lane >> 4;
  v16h b;
#pragma unroll
  for (int p = 0; p < 8; ++p) {
    int kk = ((p < 4) ? p : (p + 4)) + 4 * hi;  // pair index: k = 2*kk
    h2v t = *(const h2v*)(Bs2 + kk * 132 + col * 2);
    b[2 * p] = t[0];
    b[2 * p + 1] = t[1];
  }
  return b;
}

// ---------------------------------------------------------------------------
// Dense WMMA GEMM: C[M,N] = A[M,K](fp16,lda) @ W[N,K]^T(fp16,ldw) (+bias)(+addend)
// 256 threads, tile 128(M) x 64(N), K-step 32; each wave 32x32 = 4 WMMAs/step.
// ---------------------------------------------------------------------------
#define GBM 128
#define GBN 64
#define GBK 32

template <int OUT_HALF>
__device__ __forceinline__ void store_one(void* C, int ldc, int N,
                                          const float* bias, const float* addend,
                                          int m, int n, float v) {
  if (n >= N) return;
  if (bias) v += bias[n];
  if (addend) v += addend[(size_t)m * ldc + n];
  if (OUT_HALF) ((_Float16*)C)[(size_t)m * ldc + n] = (_Float16)v;
  else          ((float*)C)[(size_t)m * ldc + n] = v;
}

template <int OUT_HALF>
__global__ void __launch_bounds__(256)
gemm_wmma_kernel(const _Float16* __restrict__ A, int lda,
                 const _Float16* __restrict__ W, int ldw,
                 const float* __restrict__ bias,
                 const float* __restrict__ addend,
                 void* __restrict__ Cout, int ldc,
                 int M, int N, int K) {
  __shared__ _Float16 As[GBM][GBK + 8];
  __shared__ _Float16 Bs2[16 * 66 * 2];
  const int tid = threadIdx.x;
  const int m0 = blockIdx.y * GBM;
  const int n0 = blockIdx.x * GBN;
  const int wid = tid >> 5, wm = wid & 3, wn = wid >> 2;

  v8f a00 = {}, a01 = {}, a10 = {}, a11 = {};

  const int arow = tid >> 1, ahalf = tid & 1;  // A: 128 rows x 2 segs of 16 halves
  const int brow = tid >> 2, bseg = tid & 3;   // B: 64 n-rows x 4 k-segs of 8

  for (int k0 = 0; k0 < K; k0 += GBK) {
    const _Float16* ap = A + (size_t)(m0 + arow) * lda + k0 + ahalf * 16;
#if HAVE_ASYNC_LDS
    // DMA 32B per thread straight into LDS; tracked by ASYNCcnt
    async_copy16(ap,     &As[arow][ahalf * 16]);
    async_copy16(ap + 8, &As[arow][ahalf * 16 + 8]);
#else
    v8h av0 = *(const v8h*)(ap);
    v8h av1 = *(const v8h*)(ap + 8);
    *(v8h*)(&As[arow][ahalf * 16]) = av0;
    *(v8h*)(&As[arow][ahalf * 16 + 8]) = av1;
#endif

    const _Float16* wp = W + (size_t)(n0 + brow) * ldw + k0 + bseg * 8;
    v8h wv = {};
    if (n0 + brow < N) wv = *(const v8h*)(wp);
#pragma unroll
    for (int j = 0; j < 4; ++j) {
      h2v p2; p2[0] = wv[2 * j]; p2[1] = wv[2 * j + 1];
      *(h2v*)(&Bs2[(bseg * 4 + j) * 132 + brow * 2]) = p2;
    }
    if (k0 + GBK < K) {                    // global_prefetch_b8 for next tiles
      __builtin_prefetch(ap + GBK, 0, 1);
      if (n0 + brow < N) __builtin_prefetch(wp + GBK, 0, 1);
    }
#if HAVE_ASYNC_LDS
    async_wait0();
#endif
    __syncthreads();

    v16h af0 = frag_a_lds(&As[wm * 32][0], GBK + 8);
    v16h af1 = frag_a_lds(&As[wm * 32 + 16][0], GBK + 8);
    v16h bf0 = frag_b_pair(Bs2, wn * 32);
    v16h bf1 = frag_b_pair(Bs2, wn * 32 + 16);
    a00 = __builtin_amdgcn_wmma_f32_16x16x32_f16(false, af0, false, bf0, (short)0, a00, false, false);
    a01 = __builtin_amdgcn_wmma_f32_16x16x32_f16(false, af0, false, bf1, (short)0, a01, false, false);
    a10 = __builtin_amdgcn_wmma_f32_16x16x32_f16(false, af1, false, bf0, (short)0, a10, false, false);
    a11 = __builtin_amdgcn_wmma_f32_16x16x32_f16(false, af1, false, bf1, (short)0, a11, false, false);
    __syncthreads();
  }

  const int lane = tid & 31;
  const int nf = lane & 15;
  const int mb2 = (lane >> 4) * 8;
  const int mbase = m0 + wm * 32 + mb2;
  const int nbase = n0 + wn * 32 + nf;
#pragma unroll
  for (int i = 0; i < 8; ++i) {
    store_one<OUT_HALF>(Cout, ldc, N, bias, addend, mbase + i,      nbase,      a00[i]);
    store_one<OUT_HALF>(Cout, ldc, N, bias, addend, mbase + i,      nbase + 16, a01[i]);
    store_one<OUT_HALF>(Cout, ldc, N, bias, addend, mbase + 16 + i, nbase,      a10[i]);
    store_one<OUT_HALF>(Cout, ldc, N, bias, addend, mbase + 16 + i, nbase + 16, a11[i]);
  }
}

// ---------------------------------------------------------------------------
// Implicit-GEMM WMMA conv: 3x3 pad-1 conv + BN + ReLU + 2x2 pool, fused.
// Per block: conv-output tile 2(h) x 32(w), 64 output channels.
//   M = 64 spatial (m = dh*32+dw), N = 64 channels, K = 9*CI with k = r*CI+ci
//   (r = ky*3+kx); weights pre-permuted to that K-order, so (ky,kx) is scalar
//   per K-step and A-fragment addressing is a constant-stride LDS walk.
// Epilogue: 64x64 fp32 tile -> LDS -> BN+ReLU+maxpool -> fp16.
// LAYOUT 0: NCHW; LAYOUT 1: aproj[b*128+t][co*8+mel]
// ---------------------------------------------------------------------------
template <int CI, int LAYOUT>
__global__ void __launch_bounds__(256)
conv_wmma_kernel(const _Float16* __restrict__ xin,
                 const _Float16* __restrict__ wk,   // [CO][9*CI], k = r*CI+ci
                 const float* __restrict__ g, const float* __restrict__ bb,
                 const float* __restrict__ mm, const float* __restrict__ vv,
                 _Float16* __restrict__ out,
                 int H, int Wd, int CO) {
  constexpr int PW = 40;            // patch row pitch (34 used)
  constexpr int K = 9 * CI;
  __shared__ _Float16 patch[CI * 4 * PW];   // rows h0-1..h0+2, cols w0-1..w0+32
  __shared__ _Float16 Bs2[16 * 66 * 2];
  __shared__ float    Cs[64 * 68];
  const int tid = threadIdx.x;
  const int b = blockIdx.z;
  const int co0 = blockIdx.y * 64;
  const int twc = Wd >> 5;
  const int th = blockIdx.x / twc, tw = blockIdx.x - th * twc;
  const int h0 = th * 2, w0 = tw * 32;

  // stage input patch (each element read once per block, reused by all 9 taps)
  const _Float16* xb = xin + (size_t)b * CI * H * Wd;
  for (int rrow = tid; rrow < CI * 4; rrow += 256) {
    int ci = rrow >> 2, rr = rrow & 3;
    int y = h0 - 1 + rr;
    _Float16* dst = &patch[ci * 4 * PW + rr * PW];
    const _Float16* src = xb + (size_t)ci * H * Wd + (size_t)y * Wd + (w0 - 1);
    bool yok = (y >= 0) && (y < H);
    for (int c = 0; c < 34; ++c) {
      int xcol = w0 - 1 + c;
      dst[c] = (yok && xcol >= 0 && xcol < Wd) ? src[c] : (_Float16)0.f;
    }
  }

  const int wid = tid >> 5, wm = wid & 3, wn = wid >> 2;
  const int lane = tid & 31;
  const int row = lane & 15, hi = lane >> 4;
  const int m = wm * 16 + row;
  const int dh = m >> 5, dw = m & 31;
  const int brow = tid >> 2, bseg = tid & 3;

  v8f acc0 = {}, acc1 = {};
  for (int k0 = 0; k0 < K; k0 += 32) {
    // stage weight tile (paired-k layout)
    v8h wv = *(const v8h*)(wk + (size_t)(co0 + brow) * K + k0 + bseg * 8);
#pragma unroll
    for (int j = 0; j < 4; ++j) {
      h2v p2; p2[0] = wv[2 * j]; p2[1] = wv[2 * j + 1];
      *(h2v*)(&Bs2[(bseg * 4 + j) * 132 + brow * 2]) = p2;
    }
    __syncthreads();

    const int r = k0 / CI;            // CI = 32/64/128 -> shift
    const int ci0 = k0 - r * CI;
    const int ky = r / 3, kx = r - ky * 3;
    const _Float16* pb = &patch[ci0 * (4 * PW) + (dh + ky) * PW + dw + kx];
    v16h af;
#pragma unroll
    for (int p = 0; p < 8; ++p) {
      int kl = ((p < 4) ? (2 * p) : (2 * p + 8)) + 8 * hi;
      af[2 * p]     = pb[(kl)     * (4 * PW)];
      af[2 * p + 1] = pb[(kl + 1) * (4 * PW)];
    }
    v16h bf0 = frag_b_pair(Bs2, wn * 32);
    v16h bf1 = frag_b_pair(Bs2, wn * 32 + 16);
    acc0 = __builtin_amdgcn_wmma_f32_16x16x32_f16(false, af, false, bf0, (short)0, acc0, false, false);
    acc1 = __builtin_amdgcn_wmma_f32_16x16x32_f16(false, af, false, bf1, (short)0, acc1, false, false);
    __syncthreads();
  }

  // spill tile to LDS for the cross-lane pooled epilogue
  {
    int nf = lane & 15, mb2 = hi * 8;
#pragma unroll
    for (int i = 0; i < 8; ++i) {
      Cs[(wm * 16 + mb2 + i) * 68 + wn * 32 + nf]      = acc0[i];
      Cs[(wm * 16 + mb2 + i) * 68 + wn * 32 + 16 + nf] = acc1[i];
    }
  }
  __syncthreads();

  // BN + ReLU + 2x2 maxpool -> pooled row th, cols tw*16..tw*16+15
  const int n = tid & 63;
  const int co = co0 + n;
  float scale = g[co] * rsqrtf(vv[co] + LN_EPS);
  float shift = bb[co] - mm[co] * scale;
  for (int pw = tid >> 6; pw < 16; pw += 4) {
    float r0 = fmaxf(Cs[(2 * pw) * 68 + n]      * scale + shift, 0.f);
    float r1 = fmaxf(Cs[(2 * pw + 1) * 68 + n]  * scale + shift, 0.f);
    float r2 = fmaxf(Cs[(32 + 2 * pw) * 68 + n] * scale + shift, 0.f);
    float r3 = fmaxf(Cs[(33 + 2 * pw) * 68 + n] * scale + shift, 0.f);
    float best = fmaxf(fmaxf(r0, r1), fmaxf(r2, r3));
    if (LAYOUT == 0) {
      int OH = H >> 1, OW = Wd >> 1;
      out[(((size_t)b * CO + co) * OH + th) * OW + (tw * 16 + pw)] = (_Float16)best;
    } else {
      out[((size_t)b * 128 + (tw * 16 + pw)) * 2048 + co * 8 + th] = (_Float16)best;
    }
  }
}

// ---------------------------------------------------------------------------
// conv1: Cin=1 (9 MACs/output) -> memory bound, scalar direct conv.
// ---------------------------------------------------------------------------
__global__ void conv_first_kernel(const float* __restrict__ x,
                                  const float* __restrict__ w,
                                  const float* __restrict__ g,
                                  const float* __restrict__ bb,
                                  const float* __restrict__ mm,
                                  const float* __restrict__ vv,
                                  _Float16* __restrict__ out) {
  const int H = 128, Wd = 2048, OH = 64, OW = 1024, CO = 32;
  int sp = blockIdx.x * blockDim.x + threadIdx.x;
  int co = blockIdx.y, b = blockIdx.z;
  if (sp >= OH * OW) return;
  int oh = sp / OW, ow = sp % OW;
  float wl[9];
#pragma unroll
  for (int i = 0; i < 9; ++i) wl[i] = w[co * 9 + i];
  float scale = g[co] * rsqrtf(vv[co] + LN_EPS);
  float shift = bb[co] - mm[co] * scale;
  const float* xb = x + (size_t)b * H * Wd;
  float best = 0.f;
#pragma unroll
  for (int py = 0; py < 2; ++py)
#pragma unroll
    for (int px = 0; px < 2; ++px) {
      int ih = oh * 2 + py, iw = ow * 2 + px;
      float acc = 0.f;
#pragma unroll
      for (int ky = 0; ky < 3; ++ky)
#pragma unroll
        for (int kx = 0; kx < 3; ++kx) {
          int yy = ih - 1 + ky, xx = iw - 1 + kx;
          if (yy >= 0 && yy < H && xx >= 0 && xx < Wd)
            acc += xb[yy * Wd + xx] * wl[ky * 3 + kx];
        }
      best = fmaxf(best, fmaxf(acc * scale + shift, 0.f));
    }
  out[(((size_t)b * CO + co) * OH + oh) * OW + ow] = (_Float16)best;
}

// ---------------------------------------------------------------------------
// Elementwise / small kernels
// ---------------------------------------------------------------------------
__global__ void cast_f32_f16_kernel(const float* __restrict__ s,
                                    _Float16* __restrict__ d, int n) {
  int i = blockIdx.x * blockDim.x + threadIdx.x;
  if (i < n) d[i] = (_Float16)s[i];
}

// fp32 [CO][CI][3][3] -> fp16 [CO][9*CI] with k = (ky*3+kx)*CI + ci
__global__ void convw_cast_kernel(const float* __restrict__ src,
                                  _Float16* __restrict__ dst, int CO, int CI) {
  int idx = blockIdx.x * blockDim.x + threadIdx.x;
  int total = CO * CI * 9;
  if (idx >= total) return;
  int co = idx / (CI * 9);
  int rem = idx - co * CI * 9;
  int ci = rem / 9;
  int r = rem - ci * 9;
  dst[(size_t)co * CI * 9 + r * CI + ci] = (_Float16)src[idx];
}

template <int OUT_HALF>
__global__ void layernorm_kernel(const float* __restrict__ x,
                                 const float* __restrict__ g,
                                 const float* __restrict__ be,
                                 void* __restrict__ out, int D) {
  int row = blockIdx.x;
  int tid = threadIdx.x;
  __shared__ float s1[256], s2[256];
  float a = x[(size_t)row * D + tid];
  float b = x[(size_t)row * D + tid + 256];
  s1[tid] = a + b;
  s2[tid] = a * a + b * b;
  __syncthreads();
  for (int off = 128; off > 0; off >>= 1) {
    if (tid < off) { s1[tid] += s1[tid + off]; s2[tid] += s2[tid + off]; }
    __syncthreads();
  }
  float mu = s1[0] / D;
  float var = s2[0] / D - mu * mu;
  float r = rsqrtf(var + LN_EPS);
  float o0 = (a - mu) * r * g[tid] + be[tid];
  float o1 = (b - mu) * r * g[tid + 256] + be[tid + 256];
  if (OUT_HALF) {
    ((_Float16*)out)[(size_t)row * D + tid] = (_Float16)o0;
    ((_Float16*)out)[(size_t)row * D + tid + 256] = (_Float16)o1;
  } else {
    ((float*)out)[(size_t)row * D + tid] = o0;
    ((float*)out)[(size_t)row * D + tid + 256] = o1;
  }
}

__global__ void dwconv_silu_kernel(const float* __restrict__ xz,
                                   const float* __restrict__ cw,
                                   const float* __restrict__ cb,
                                   _Float16* __restrict__ xc) {
  int idx = blockIdx.x * blockDim.x + threadIdx.x;
  if (idx >= 8 * 128 * 1024) return;
  int d = idx & 1023;
  int t = (idx >> 10) & 127;
  int b = idx >> 17;
  float s = cb[d];
#pragma unroll
  for (int j = 0; j < 3; ++j) {
    int tt = t + j - 2;
    if (tt >= 0) s += xz[(size_t)(b * 128 + tt) * 2048 + d] * cw[d * 3 + j];
  }
  s = s / (1.f + __expf(-s));
  xc[(size_t)(b * 128 + t) * 1024 + d] = (_Float16)s;
}

__global__ void mamba_scan_kernel(const float* __restrict__ dtlin,
                                  const _Float16* __restrict__ xc,
                                  const _Float16* __restrict__ dbc,
                                  const float* __restrict__ xz,
                                  const float* __restrict__ A_log,
                                  const float* __restrict__ Dp,
                                  _Float16* __restrict__ yout) {
  int idx = blockIdx.x * blockDim.x + threadIdx.x;  // 8192 = (b,d)
  int b = idx >> 10, d = idx & 1023;
  float a[64], h[64];
#pragma unroll
  for (int n = 0; n < 64; ++n) {
    a[n] = -__expf(A_log[d * 64 + n]);
    h[n] = 0.f;
  }
  float Dd = Dp[d];
  for (int t = 0; t < 128; ++t) {
    size_t r = (size_t)b * 128 + t;
    float dtv = dtlin[r * 1024 + d];
    dtv = (dtv > 20.f) ? dtv : log1pf(__expf(dtv));
    float xv = (float)xc[r * 1024 + d];
    float dx = dtv * xv;
    float y = 0.f;
#pragma unroll
    for (int n = 0; n < 64; ++n) {
      float Bv = (float)dbc[r * 160 + 32 + n];
      float Cv = (float)dbc[r * 160 + 96 + n];
      float dA = __expf(dtv * a[n]);
      h[n] = dA * h[n] + dx * Bv;
      y += h[n] * Cv;
    }
    float zv = xz[r * 2048 + 1024 + d];
    float yv = (y + xv * Dd) * (zv / (1.f + __expf(-zv)));
    yout[r * 1024 + d] = (_Float16)yv;
  }
}

__global__ void head_kernel(const float* __restrict__ xf,
                            const float* __restrict__ w1, const float* __restrict__ b1,
                            const float* __restrict__ w2, const float* __restrict__ b2,
                            const float* __restrict__ w3, const float* __restrict__ b3,
                            float* __restrict__ out) {
  int b = blockIdx.x;
  int tid = threadIdx.x;
  __shared__ float mean[512];
  __shared__ float h1[128];
  __shared__ float h2s[64];
  for (int c = tid; c < 512; c += 256) {
    float s = 0.f;
    for (int t = 0; t < 128; ++t) s += xf[(size_t)(b * 128 + t) * 512 + c];
    mean[c] = s * (1.f / 128.f);
  }
  __syncthreads();
  for (int i = tid; i < 128; i += 256) {
    float s = b1[i];
    for (int k = 0; k < 512; ++k) s += mean[k] * w1[(size_t)i * 512 + k];
    h1[i] = fmaxf(s, 0.f);
  }
  __syncthreads();
  if (tid < 64) {
    float s = b2[tid];
    for (int k = 0; k < 128; ++k) s += h1[k] * w2[(size_t)tid * 128 + k];
    h2s[tid] = fmaxf(s, 0.f);
  }
  __syncthreads();
  if (tid < 2) {
    float s = b3[tid];
    for (int k = 0; k < 64; ++k) s += h2s[k] * w3[(size_t)tid * 64 + k];
    out[b * 2 + tid] = s;
  }
}

// ---------------------------------------------------------------------------
// Host launch
// ---------------------------------------------------------------------------
static void launch_gemm(const _Float16* A, int lda, const _Float16* W, int ldw,
                        const float* bias, const float* addend, void* C, int ldc,
                        int M, int N, int K, bool outHalf, hipStream_t s) {
  dim3 grid((N + GBN - 1) / GBN, (M + GBM - 1) / GBM);
  if (outHalf)
    gemm_wmma_kernel<1><<<grid, 256, 0, s>>>(A, lda, W, ldw, bias, addend, C, ldc, M, N, K);
  else
    gemm_wmma_kernel<0><<<grid, 256, 0, s>>>(A, lda, W, ldw, bias, addend, C, ldc, M, N, K);
}

extern "C" void kernel_launch(void* const* d_in, const int* in_sizes, int n_in,
                              void* d_out, int out_size, void* d_ws, size_t ws_size,
                              hipStream_t stream) {
  (void)in_sizes; (void)n_in; (void)out_size; (void)ws_size;
  auto F = [&](int i) { return (const float*)d_in[i]; };

  uint8_t* base = (uint8_t*)d_ws;
  size_t off = 0;
  auto alloc = [&](size_t bytes) -> void* {
    off = (off + 255) & ~(size_t)255;
    void* p = base + off;
    off += bytes;
    return p;
  };

  // ---- conv chain region (peak workspace; later buffers reuse the front) ----
  h16* c1h  = (h16*)alloc((size_t)8 * 32 * 64 * 1024 * 2);   // 33.5 MB
  h16* c2h  = (h16*)alloc((size_t)8 * 64 * 32 * 512 * 2);    // 16.8 MB
  h16* c3h  = (h16*)alloc((size_t)8 * 128 * 16 * 256 * 2);   //  8.4 MB
  h16* c2wh = (h16*)alloc((size_t)64 * 288 * 2);             // conv weights fp16
  h16* c3wh = (h16*)alloc((size_t)128 * 576 * 2);            // (permuted K-order)
  h16* c4wh = (h16*)alloc((size_t)256 * 1152 * 2);

  convw_cast_kernel<<<(64 * 288 + 255) / 256, 256, 0, stream>>>(F(6), c2wh, 64, 32);
  convw_cast_kernel<<<(128 * 576 + 255) / 256, 256, 0, stream>>>(F(11), c3wh, 128, 64);
  convw_cast_kernel<<<(256 * 1152 + 255) / 256, 256, 0, stream>>>(F(16), c4wh, 256, 128);

  conv_first_kernel<<<dim3(256, 32, 8), 256, 0, stream>>>(
      F(0), F(1), F(2), F(3), F(4), F(5), c1h);
  // conv2: [8,32,64,1024] -> [8,64,32,512]
  conv_wmma_kernel<32, 0><<<dim3(32 * 32, 1, 8), 256, 0, stream>>>(
      c1h, c2wh, F(7), F(8), F(9), F(10), c2h, 64, 1024, 64);
  // conv3: [8,64,32,512] -> [8,128,16,256]
  conv_wmma_kernel<64, 0><<<dim3(16 * 16, 2, 8), 256, 0, stream>>>(
      c2h, c3wh, F(12), F(13), F(14), F(15), c3h, 32, 512, 128);

  // ---- reuse region from workspace start (stays < 50 MB => clear of c3h) ----
  off = 0;
  h16* aproj = (h16*)alloc((size_t)1024 * 2048 * 2);
  // conv4: [8,128,16,256] -> aproj[b*128+t][co*8+mel]
  conv_wmma_kernel<128, 1><<<dim3(8 * 8, 4, 8), 256, 0, stream>>>(
      c3h, c4wh, F(17), F(18), F(19), F(20), aproj, 16, 256, 256);

  auto cast = [&](const float* s, h16* d, int n) {
    cast_f32_f16_kernel<<<(n + 255) / 256, 256, 0, stream>>>(s, d, n);
  };
  h16* projwh = (h16*)alloc((size_t)512 * 2048 * 2);
  cast(F(21), projwh, 512 * 2048);
  h16 *inwh[2], *xprojwh[2], *dtwh[2], *outwh[2];
  for (int blk = 0; blk < 2; ++blk) {
    int pb = 23 + blk * 11;
    inwh[blk]    = (h16*)alloc((size_t)2048 * 512 * 2);
    xprojwh[blk] = (h16*)alloc((size_t)160 * 1024 * 2);
    dtwh[blk]    = (h16*)alloc((size_t)1024 * 32 * 2);
    outwh[blk]   = (h16*)alloc((size_t)512 * 1024 * 2);
    cast(F(pb + 2),  inwh[blk],    2048 * 512);
    cast(F(pb + 5),  xprojwh[blk], 160 * 1024);
    cast(F(pb + 6),  dtwh[blk],    1024 * 32);
    cast(F(pb + 10), outwh[blk],   512 * 1024);
  }

  float* xA    = (float*)alloc((size_t)1024 * 512 * 4);
  float* xB    = (float*)alloc((size_t)1024 * 512 * 4);
  h16*   xlnh  = (h16*)alloc((size_t)1024 * 512 * 2);
  float* xz    = (float*)alloc((size_t)1024 * 2048 * 4);
  h16*   xch   = (h16*)alloc((size_t)1024 * 1024 * 2);
  h16*   dbch  = (h16*)alloc((size_t)1024 * 160 * 2);
  float* dtlin = (float*)alloc((size_t)1024 * 1024 * 4);
  h16*   yh    = (h16*)alloc((size_t)1024 * 1024 * 2);
  float* xf    = (float*)alloc((size_t)1024 * 512 * 4);

  // projection: x = aproj @ proj_w.T + proj_b   [1024,512]
  launch_gemm(aproj, 2048, projwh, 2048, F(22), nullptr, xA, 512,
              1024, 512, 2048, false, stream);

  float* xcur = xA;
  float* xnext = xB;
  for (int blk = 0; blk < 2; ++blk) {
    int pb = 23 + blk * 11;
    layernorm_kernel<1><<<1024, 256, 0, stream>>>(xcur, F(pb + 0), F(pb + 1), xlnh, 512);
    launch_gemm(xlnh, 512, inwh[blk], 512, nullptr, nullptr, xz, 2048,
                1024, 2048, 512, false, stream);
    dwconv_silu_kernel<<<4096, 256, 0, stream>>>(xz, F(pb + 3), F(pb + 4), xch);
    launch_gemm(xch, 1024, xprojwh[blk], 1024, nullptr, nullptr, dbch, 160,
                1024, 160, 1024, true, stream);
    launch_gemm(dbch, 160, dtwh[blk], 32, F(pb + 7), nullptr, dtlin, 1024,
                1024, 1024, 32, false, stream);
    mamba_scan_kernel<<<32, 256, 0, stream>>>(dtlin, xch, dbch, xz,
                                              F(pb + 8), F(pb + 9), yh);
    launch_gemm(yh, 1024, outwh[blk], 1024, nullptr, xcur, xnext, 512,
                1024, 512, 1024, false, stream);
    float* t = xcur; xcur = xnext; xnext = t;
  }

  layernorm_kernel<0><<<1024, 256, 0, stream>>>(xcur, F(45), F(46), xf, 512);
  head_kernel<<<8, 256, 0, stream>>>(xf, F(47), F(48), F(49), F(50), F(51), F(52),
                                     (float*)d_out);
}
